// VectorizedAttentionHead_30777735643862
// MI455X (gfx1250) — compile-verified, matrix operand order checked by reading, not run on hardware
//
#include <hip/hip_runtime.h>

// Problem constants (match reference)
#define BB 4
#define SS 2048
#define EE 2048
#define HH 16
#define DD 128

typedef __attribute__((ext_vector_type(16))) __bf16 v16bf;
typedef __attribute__((ext_vector_type(8)))  __bf16 bf16x8;
typedef __attribute__((ext_vector_type(4)))  __bf16 bf16x4;
typedef __attribute__((ext_vector_type(8)))  float  v8f;
typedef __attribute__((ext_vector_type(4)))  unsigned int u32x4;
typedef __attribute__((ext_vector_type(8)))  int  i32x8;
typedef __attribute__((ext_vector_type(4)))  int  i32x4;

static __device__ __forceinline__ __bf16 f2bf(float f) {
  union { float f; unsigned u; } a; a.f = f;
  unsigned u = a.u + 0x7FFFu + ((a.u >> 16) & 1u);   // round-to-nearest-even
  union { unsigned short s; __bf16 b; } b; b.s = (unsigned short)(u >> 16);
  return b.b;
}

// NT fragment loader for V_WMMA_F32_16X16X32_BF16.  Source is row-major
// [rows, ld] bf16 (global or LDS); fragment holds 16 K-values of one row per
// lane, split per ISA 7.12.2: lane<16 -> K {k0..k0+7, k0+16..k0+23},
// lane>=16 -> K {k0+8..k0+15, k0+24..k0+31}.
static __device__ __forceinline__ v16bf load_frag_nt(const __bf16* p,
                                                     int row, int ld, int k0, int lane) {
  const __bf16* base = p + (size_t)row * ld + k0 + ((lane & 16) ? 8 : 0);
  bf16x8 lo = *(const bf16x8*)base;          // 16B b128 load
  bf16x8 hi = *(const bf16x8*)(base + 16);   // 16B b128 load
  return __builtin_shufflevector(lo, hi, 0, 1, 2, 3, 4, 5, 6, 7,
                                         8, 9, 10, 11, 12, 13, 14, 15);
}

static __device__ __forceinline__ v8f wmma_bf16(v16bf a, v16bf b, v8f c) {
  return __builtin_amdgcn_wmma_f32_16x16x32_bf16(false, a, false, b,
                                                 (short)0, c, false, false);
}

// Low 32 bits of a generic pointer to a __shared__ object = LDS byte offset
// (ISA 10.2: LDS aperture address is {aperture_hi, offset32}).
static __device__ __forceinline__ unsigned lds_off(const void* p) {
  return (unsigned)(uintptr_t)p;
}

// ---------------------------------------------------------------------------
// TDM: async 2D tile load global -> LDS (bf16 elements).
// Builds a D# per ISA 8.3-8.6: group0 = {flags, lds_addr, global_addr, type=2},
// group1 = {mask/data_size, tensor_dim0/1, tile_dim0/1 (tile_dim2=0 -> 2D),
// tensor_dim0_stride}.  Tracked by TENSORcnt.
// ---------------------------------------------------------------------------
static __device__ __forceinline__ void tdm_load_2d(unsigned lds_byte_off,
                                                   const void* gaddr,
                                                   unsigned tile_x,    // contiguous elems
                                                   unsigned tile_y,    // rows
                                                   unsigned stride_x)  // row stride (elems)
{
  unsigned long long ga = (unsigned long long)(uintptr_t)gaddr;
  u32x4 g0;
  g0[0] = 1u;                                     // count=1, user mode, no gather
  g0[1] = lds_byte_off;                           // D#.lds_addr (bytes)
  g0[2] = (unsigned)(ga & 0xFFFFFFFFu);           // global_addr[31:0]
  g0[3] = (unsigned)((ga >> 32) & 0x01FFFFFFu)    // global_addr[56:32]
        | 0x80000000u;                            // type=2 ("image")
  i32x8 g1;
  g1[0] = 0x10000;                                // wg_mask=0, data_size=1 (2 bytes)
  g1[1] = (int)((tile_x & 0xFFFFu) << 16);        // tensor_dim0[15:0]  @ bits 63:48
  g1[2] = (int)((tile_x >> 16) & 0xFFFFu)         // tensor_dim0[31:16] @ bits 79:64
        | (int)((tile_y & 0xFFFFu) << 16);        // tensor_dim1[15:0]  @ bits 95:80
  g1[3] = (int)((tile_y >> 16) & 0xFFFFu)         // tensor_dim1[31:16] @ bits 111:96
        | (int)(tile_x << 16);                    // tile_dim0          @ bits 127:112
  g1[4] = (int)(tile_y & 0xFFFFu);                // tile_dim1 (tile_dim2=0 -> 2D)
  g1[5] = (int)stride_x;                          // tensor_dim0_stride[31:0]
  g1[6] = 0;                                      // stride0[47:32]=0, dim1_stride lo=0
  g1[7] = 0;
  i32x4 gz = {0, 0, 0, 0};
#if __has_include(<hip/amd_detail/amd_gfx1250_TDM.h>)
  i32x8 gz8 = {0, 0, 0, 0, 0, 0, 0, 0};
  __builtin_amdgcn_tensor_load_to_lds(g0, g1, gz, gz, gz8, 0);
#else
  __builtin_amdgcn_tensor_load_to_lds(g0, g1, gz, gz, 0);
#endif
}

// ---------------------------------------------------------------------------
// Kernel 0: f32 -> bf16 cast (vectorized 4-wide)
// ---------------------------------------------------------------------------
__global__ __launch_bounds__(256) void cast_f32_to_bf16(const float* __restrict__ in,
                                                        __bf16* __restrict__ out,
                                                        int n4) {
  int i = blockIdx.x * 256 + threadIdx.x;
  if (i >= n4) return;
  float4 f = ((const float4*)in)[i];
  bf16x4 o;
  o[0] = f2bf(f.x); o[1] = f2bf(f.y); o[2] = f2bf(f.z); o[3] = f2bf(f.w);
  ((bf16x4*)out)[i] = o;
}

// ---------------------------------------------------------------------------
// Kernel 1: Y = X @ W^T  (NT GEMM, bf16 in / f32 acc / bf16 out)
//   layout 0: Y as [B,H,S,D] (Q, K);  layout 1: Y as [B,H,D,S] (V transposed)
// Block = 256 threads = 8 waves in 2(M)x4(N); wave tile 32x32 (2x2 WMMA).
// ---------------------------------------------------------------------------
__global__ __launch_bounds__(256) void qkv_gemm(const __bf16* __restrict__ X,
                                                const __bf16* __restrict__ W,
                                                __bf16* __restrict__ Y,
                                                int layout) {
  const int lane   = threadIdx.x & 31;
  const int wave   = threadIdx.x >> 5;
  const int m_base = blockIdx.x * 64  + (wave >> 2) * 32;
  const int n_base = blockIdx.y * 128 + (wave & 3)  * 32;

  v8f acc[2][2] = {};

  for (int k0 = 0; k0 < EE; k0 += 32) {
    if (k0 + 64 < EE) {  // pull next chunks toward the WGP (global_prefetch_b8)
      __builtin_prefetch(X + (size_t)(m_base + (lane & 15)) * EE + k0 + 64, 0, 0);
      __builtin_prefetch(W + (size_t)(n_base + (lane & 15)) * EE + k0 + 64, 0, 0);
    }
    v16bf af[2], bfg[2];
#pragma unroll
    for (int t = 0; t < 2; ++t) {
      af[t]  = load_frag_nt(X, m_base + t * 16 + (lane & 15), EE, k0, lane);
      bfg[t] = load_frag_nt(W, n_base + t * 16 + (lane & 15), EE, k0, lane);
    }
#pragma unroll
    for (int tm = 0; tm < 2; ++tm)
#pragma unroll
      for (int tn = 0; tn < 2; ++tn)
        acc[tm][tn] = wmma_bf16(af[tm], bfg[tn], acc[tm][tn]);
  }

  // C layout: VGPR r, lanes 0-15 -> (M=r, N=lane); lanes 16-31 -> (M=8+r).
  const int h = lane >> 4, nl = lane & 15;
  if (layout == 0) {
#pragma unroll
    for (int tm = 0; tm < 2; ++tm)
#pragma unroll
      for (int tn = 0; tn < 2; ++tn)
#pragma unroll
        for (int r = 0; r < 8; ++r) {
          int m = m_base + tm * 16 + h * 8 + r;   // m = b*S + s
          int n = n_base + tn * 16 + nl;          // n = head*D + d
          int b  = m >> 11, s = m & (SS - 1);
          int hd = n >> 7,  d = n & (DD - 1);
          Y[((size_t)(b * HH + hd) * SS + s) * DD + d] = f2bf(acc[tm][tn][r]);
        }
  } else {
#pragma unroll
    for (int tm = 0; tm < 2; ++tm)
#pragma unroll
      for (int tn = 0; tn < 2; ++tn)
#pragma unroll
        for (int r = 0; r < 8; ++r) {
          int m = m_base + tm * 16 + h * 8 + r;
          int n = n_base + tn * 16 + nl;
          int b  = m >> 11, s = m & (SS - 1);
          int hd = n >> 7,  d = n & (DD - 1);
          Y[((size_t)(b * HH + hd) * DD + d) * SS + s] = f2bf(acc[tm][tn][r]);
        }
  }
}

// ---------------------------------------------------------------------------
// Kernel 2: causal flash attention forward with TDM-staged K/V tiles.
//   Q,K: bf16 [B*H, S, D];  Vt: bf16 [B*H, D, S];  O: f32 [B*H, S, D]
// Block = 8 waves = 128 query rows of one (b,h); wave w owns rows Q0+16w..+15.
// KV loop in 32-key steps; K (32x128) and V (128x32) tiles are DMA'd into
// double-buffered LDS by the Tensor Data Mover (issued by wave 0), so the
// load of step jt+1 overlaps compute of step jt.  8 waves share each tile.
// ---------------------------------------------------------------------------
__global__ __launch_bounds__(256) void flash_fwd(const __bf16* __restrict__ Q,
                                                 const __bf16* __restrict__ Km,
                                                 const __bf16* __restrict__ Vt,
                                                 float* __restrict__ O) {
  __shared__ __bf16 kTile[2][32 * 128];   // 8 KB per stage
  __shared__ __bf16 vTile[2][128 * 32];   // 8 KB per stage
  __shared__ float  pScr[8][16 * 32];     // per-wave P staging (2 KB each)

  const int lane = threadIdx.x & 31;
  const int wave = threadIdx.x >> 5;
  float* pl = &pScr[wave][0];

  const int bh = blockIdx.x >> 4;          // 16 row-chunks of 128 per (b,h)
  const int Q0 = (blockIdx.x & 15) * 128;
  const int q0 = Q0 + wave * 16;

  const __bf16* Qb = Q  + (size_t)bh * SS * DD;
  const __bf16* Kb = Km + (size_t)bh * SS * DD;
  const __bf16* Vb = Vt + (size_t)bh * DD * SS;

  const int   h  = lane >> 4, nl = lane & 15;
  const float scale   = 0.08838834764831845f;  // 1/sqrt(128)
  const float NEG_BIG = -1e30f;

  // Q fragments held in registers for the whole KV loop (4 d-chunks of 32).
  v16bf qf[4];
#pragma unroll
  for (int kc = 0; kc < 4; ++kc)
    qf[kc] = load_frag_nt(Qb, q0 + (lane & 15), DD, kc * 32, lane);

  v8f o[8] = {};
  float mst[8], lst[8];
#pragma unroll
  for (int r = 0; r < 8; ++r) { mst[r] = NEG_BIG; lst[r] = 0.f; }

  const int jt_w   = (q0 + 15) >> 5;    // last KV step this wave needs
  const int jt_blk = (Q0 + 127) >> 5;   // last KV step any wave in block needs

  // Prologue: stage step 0.
  if (wave == 0) {
    tdm_load_2d(lds_off(&kTile[0][0]), Kb, 128, 32, 128);
    tdm_load_2d(lds_off(&vTile[0][0]), Vb, 32, 128, SS);
  }

  for (int jt = 0; jt <= jt_blk; ++jt) {
    const int cur = jt & 1;
    if (wave == 0)
      __builtin_amdgcn_s_wait_tensorcnt(0);   // current tiles landed in LDS
    __syncthreads();                          // visible to all waves; previous
                                              // buffer no longer being read
    if (wave == 0 && jt < jt_blk) {           // overlap next DMA with compute
      const int jn = (jt + 1) * 32;
      tdm_load_2d(lds_off(&kTile[cur ^ 1][0]), Kb + (size_t)jn * DD, 128, 32, 128);
      tdm_load_2d(lds_off(&vTile[cur ^ 1][0]), Vb + jn, 32, 128, SS);
    }

    if (jt <= jt_w) {
      const int j0 = jt * 32;

      // scores S = Q @ K^T  (2 n-tiles x 4 k-chunks = 8 WMMAs), K from LDS
      v8f sa[2] = {};
#pragma unroll
      for (int kc = 0; kc < 4; ++kc) {
#pragma unroll
        for (int tn = 0; tn < 2; ++tn) {
          v16bf kf = load_frag_nt(&kTile[cur][0], tn * 16 + (lane & 15),
                                  128, kc * 32, lane);
          sa[tn] = wmma_bf16(qf[kc], kf, sa[tn]);
        }
      }

      // scale + causal mask (only this wave's diagonal step needs masking)
      const bool diag = (jt == jt_w);
#pragma unroll
      for (int tn = 0; tn < 2; ++tn)
#pragma unroll
        for (int r = 0; r < 8; ++r) {
          float v = sa[tn][r] * scale;
          if (diag) {
            int qi = q0 + h * 8 + r;
            int kj = j0 + tn * 16 + nl;
            if (kj > qi) v = NEG_BIG;
          }
          sa[tn][r] = v;
        }

      // online softmax: row stats via 16-lane shfl reductions
#pragma unroll
      for (int r = 0; r < 8; ++r) {
        float mx = fmaxf(sa[0][r], sa[1][r]);
#pragma unroll
        for (int off = 1; off < 16; off <<= 1)
          mx = fmaxf(mx, __shfl_xor(mx, off, 32));
        float mnew  = fmaxf(mst[r], mx);
        float alpha = __expf(mst[r] - mnew);
        float p0 = __expf(sa[0][r] - mnew);
        float p1 = __expf(sa[1][r] - mnew);
        sa[0][r] = p0; sa[1][r] = p1;
        float rs = p0 + p1;
#pragma unroll
        for (int off = 1; off < 16; off <<= 1)
          rs += __shfl_xor(rs, off, 32);
        lst[r] = lst[r] * alpha + rs;
        mst[r] = mnew;
#pragma unroll
        for (int td = 0; td < 8; ++td)
          o[td][r] *= alpha;
      }

      // stage P through wave-private LDS: C-layout -> row-major [16][32]
#pragma unroll
      for (int tn = 0; tn < 2; ++tn)
#pragma unroll
        for (int r = 0; r < 8; ++r)
          pl[(h * 8 + r) * 32 + tn * 16 + nl] = sa[tn][r];
      // per-wave LDS ops are in-order; rebuild P as an A-fragment (bf16)
      const float* prow = pl + (lane & 15) * 32 + ((lane & 16) ? 8 : 0);
      v16bf pf;
#pragma unroll
      for (int e = 0; e < 8; ++e) {
        pf[e]     = f2bf(prow[e]);
        pf[e + 8] = f2bf(prow[e + 16]);
      }

      // O += P @ V  (8 d-tiles, K=32 keys per WMMA), V from LDS
#pragma unroll
      for (int td = 0; td < 8; ++td) {
        v16bf vf = load_frag_nt(&vTile[cur][0], td * 16 + (lane & 15),
                                32, 0, lane);
        o[td] = wmma_bf16(pf, vf, o[td]);
      }
    }
  }

  // epilogue: divide by row sum, store f32 [B,H,S,D]
  float* Ob = O + (size_t)bh * SS * DD;
#pragma unroll
  for (int td = 0; td < 8; ++td)
#pragma unroll
    for (int r = 0; r < 8; ++r)
      Ob[(size_t)(q0 + h * 8 + r) * DD + td * 16 + nl] = o[td][r] / lst[r];
}

// ---------------------------------------------------------------------------
// Launch
// ---------------------------------------------------------------------------
extern "C" void kernel_launch(void* const* d_in, const int* in_sizes, int n_in,
                              void* d_out, int out_size, void* d_ws, size_t ws_size,
                              hipStream_t stream) {
  (void)in_sizes; (void)n_in; (void)out_size; (void)ws_size;
  const float* x  = (const float*)d_in[0];
  const float* Wq = (const float*)d_in[1];
  const float* Wk = (const float*)d_in[2];
  const float* Wv = (const float*)d_in[3];
  float* out = (float*)d_out;

  const size_t nx = (size_t)BB * SS * EE;  // 16,777,216 elements
  const size_t nw = (size_t)EE * EE;       //  4,194,304 elements

  char* ws = (char*)d_ws;                  // needs ~152 MB total
  __bf16* xb  = (__bf16*)ws;
  __bf16* wqb = xb  + nx;
  __bf16* wkb = wqb + nw;
  __bf16* wvb = wkb + nw;
  __bf16* Qb  = wvb + nw;                  // [B,H,S,D]
  __bf16* Kb  = Qb  + nx;                  // [B,H,S,D]
  __bf16* Vtb = Kb  + nx;                  // [B,H,D,S]

  cast_f32_to_bf16<<<(int)(nx / 4 / 256), 256, 0, stream>>>(x,  xb,  (int)(nx / 4));
  cast_f32_to_bf16<<<(int)(nw / 4 / 256), 256, 0, stream>>>(Wq, wqb, (int)(nw / 4));
  cast_f32_to_bf16<<<(int)(nw / 4 / 256), 256, 0, stream>>>(Wk, wkb, (int)(nw / 4));
  cast_f32_to_bf16<<<(int)(nw / 4 / 256), 256, 0, stream>>>(Wv, wvb, (int)(nw / 4));

  dim3 ggrid(BB * SS / 64, EE / 128);      // (128, 16)
  qkv_gemm<<<ggrid, 256, 0, stream>>>(xb, wqb, Qb,  0);
  qkv_gemm<<<ggrid, 256, 0, stream>>>(xb, wkb, Kb,  0);
  qkv_gemm<<<ggrid, 256, 0, stream>>>(xb, wvb, Vtb, 1);

  // one block per (b,h, 128-row query chunk): 4*16*16 = 1024 blocks
  flash_fwd<<<BB * HH * (SS / 128), 256, 0, stream>>>(Qb, Kb, Vtb, out);
}
